// MeshUpMP_62947040690379
// MI455X (gfx1250) — compile-verified, matrix operand order checked by reading, not run on hardware
//
#include <hip/hip_runtime.h>
#include <stdint.h>

// ---------- constants ----------
#define F_IN   256
#define F_W    512
#define IN_MSG 1024
#define TILE_M 64
#define PAD    16
#define LDA    (IN_MSG + PAD)   // 1040 bf16 elems per msg row
#define LDB    (F_W + PAD)      // 528  bf16 elems per act row
#define LN_EPS 1e-5f

typedef __bf16 v16bf __attribute__((ext_vector_type(16)));
typedef float  v8f   __attribute__((ext_vector_type(8)));

union FragBF {
    v16bf bf;
    uint4 q[2];
};

__device__ __forceinline__ unsigned short f32_to_bf16(float f) {
    unsigned int u = __float_as_uint(f);
    u += 0x7FFFu + ((u >> 16) & 1u);        // round-to-nearest-even
    return (unsigned short)(u >> 16);
}

__device__ __forceinline__ float selu(float x) {
    const float kS = 1.0507009873554805f;
    const float kA = 1.6732632423543772f;
    return x > 0.0f ? kS * x : kS * kA * (__expf(x) - 1.0f);
}

// ---------- weight f32 -> bf16 prep ----------
__global__ void cvt_bf16_kernel(const float* __restrict__ src,
                                unsigned short* __restrict__ dst, int n) {
    int i = blockIdx.x * blockDim.x + threadIdx.x;
    if (i < n) dst[i] = f32_to_bf16(src[i]);
}

// ---------- one dense layer: dst = selu(src @ W^T + b) ----------
// src: LDS bf16 [64][ld_src]; W: global bf16 [F_W][K] row-major; b: f32.
// K-outer / N-inner: a wave keeps ALL 16 accumulator tiles of its 256-col
// half live (128 VGPRs); each k-step loads ONE A fragment from LDS and 16
// independent B pairs from global (immediate offsets off one base address),
// feeding 16 back-to-back WMMAs — deep load/WMMA overlap without unrolling K.
template <bool TO_LDS, int K>
__device__ __forceinline__ void gemm_layer(
    const unsigned short* __restrict__ Wbf,
    const float* __restrict__ bias,
    const unsigned short* __restrict__ src, int ld_src,
    unsigned short* dstL, int ld_dst,
    float* dstG, long long row0, long long n_hr,
    int lane, int wave)
{
    const int s  = wave >> 1;        // 4 row strips of 16
    const int h  = wave & 1;         // 2 column halves of 256
    const int nl = lane & 15;        // N (B/D) or M (A) within tile
    const int kg = lane >> 4;        // K-group select
    const int rbase = s * 16;
    const int ncol0 = h * 256;

    v8f acc[16];
    #pragma unroll
    for (int c = 0; c < 16; ++c) {
        const float bv = bias[ncol0 + c * 16 + nl];
        acc[c] = (v8f){ bv, bv, bv, bv, bv, bv, bv, bv };
    }

    const unsigned short* ap = src + (rbase + nl) * ld_src + kg * 8;
    const unsigned short* bp = Wbf + (long long)(ncol0 + nl) * K + kg * 8;

    #pragma unroll 1                 // keep K rolled: no whole-panel hoisting
    for (int kt = 0; kt < K; kt += 32) {
        FragBF a;
        a.q[0] = *(const uint4*)(ap + kt);
        a.q[1] = *(const uint4*)(ap + kt + 16);
        #pragma unroll
        for (int c = 0; c < 16; ++c) {
            FragBF b;
            // column tile c: constant byte offset c*16*K*2 (fits 24-bit IOFFSET)
            b.q[0] = *(const uint4*)(bp + c * 16 * K + kt);
            b.q[1] = *(const uint4*)(bp + c * 16 * K + kt + 16);
            acc[c] = __builtin_amdgcn_wmma_f32_16x16x32_bf16(
                false, a.bf, false, b.bf, (short)0, acc[c], false, false);
        }
    }

    #pragma unroll
    for (int c = 0; c < 16; ++c) {
        const int n0 = ncol0 + c * 16;
        #pragma unroll
        for (int i = 0; i < 8; ++i) {
            const int m = i + 8 * kg;        // D layout: VGPR i, lane half
            const float y = selu(acc[c][i]);
            if (TO_LDS) {
                dstL[(rbase + m) * ld_dst + n0 + nl] = f32_to_bf16(y);
            } else {
                const long long gr = row0 + rbase + m;
                if (gr < n_hr) dstG[gr * F_W + n0 + nl] = y;
            }
        }
    }
}

// ---------- fused kernel: msg build + LayerNorm + 4x (Linear+SELU) ----------
__global__ __launch_bounds__(256) void meshup_main(
    const float* __restrict__ v,
    const float* __restrict__ e_rel,
    const int*   __restrict__ idx,
    const float* __restrict__ v_skip,
    const float* __restrict__ We,
    const float* __restrict__ be,
    const float* __restrict__ ln_g,
    const float* __restrict__ ln_b,
    const unsigned short* __restrict__ Wb0, const float* __restrict__ b0,
    const unsigned short* __restrict__ Wb1, const float* __restrict__ b1,
    const unsigned short* __restrict__ Wb2, const float* __restrict__ b2,
    const unsigned short* __restrict__ Wb3, const float* __restrict__ b3,
    float* __restrict__ out,
    long long n_hr)
{
    extern __shared__ char smem_raw[];
    unsigned short* bufA = (unsigned short*)smem_raw;          // [64][LDA] msg
    unsigned short* bufB = bufA + TILE_M * LDA;                // [64][LDB]
    unsigned short* bufC = bufB + TILE_M * LDB;                // [64][LDB]

    const int lane = threadIdx.x & 31;
    const int wave = threadIdx.x >> 5;
    const long long row0 = (long long)blockIdx.x * TILE_M;

    // ---- Stage 1: build msg row, LayerNorm, write bf16 to LDS ----
    // wave handles 8 rows; 32 lanes x 32 elems cover IN_MSG=1024
    for (int i = 0; i < 8; ++i) {
        const int r = wave * 8 + i;
        const long long grow = row0 + r;
        unsigned short* rowp = bufA + r * LDA;

        if (grow < n_hr) {
            const float er = e_rel[grow];
            const long long vsrc = (long long)idx[grow] * F_IN;
            float vals[32];
            float s = 0.f, s2 = 0.f;
            #pragma unroll
            for (int t = 0; t < 32; ++t) {
                const int j = t * 32 + lane;
                float x;
                if (j < F_W)                x = fmaf(-er, We[j], be[j]);       // edge enc
                else if (j < F_W + F_IN)    x = v[vsrc + (j - F_W)];           // gather
                else                        x = v_skip[grow * F_IN + (j - F_W - F_IN)];
                vals[t] = x; s += x; s2 += x * x;
            }
            // wave32 tree reduction
            #pragma unroll
            for (int m = 16; m >= 1; m >>= 1) {
                s  += __shfl_xor(s,  m, 32);
                s2 += __shfl_xor(s2, m, 32);
            }
            const float mu   = s  * (1.0f / IN_MSG);
            const float var  = s2 * (1.0f / IN_MSG) - mu * mu;
            const float rstd = rsqrtf(var + LN_EPS);
            #pragma unroll
            for (int t = 0; t < 32; ++t) {
                const int j = t * 32 + lane;
                const float y = (vals[t] - mu) * rstd * ln_g[j] + ln_b[j];
                rowp[j] = f32_to_bf16(y);
            }
        } else {
            #pragma unroll
            for (int t = 0; t < 32; ++t) rowp[t * 32 + lane] = 0;
        }
    }
    __syncthreads();

    // ---- Stage 2..5: four WMMA GEMM layers ----
    gemm_layer<true,  IN_MSG>(Wb0, b0, bufA, LDA, bufB, LDB, nullptr, 0, 0, lane, wave);
    __syncthreads();
    gemm_layer<true,  F_W   >(Wb1, b1, bufB, LDB, bufC, LDB, nullptr, 0, 0, lane, wave);
    __syncthreads();
    gemm_layer<true,  F_W   >(Wb2, b2, bufC, LDB, bufB, LDB, nullptr, 0, 0, lane, wave);
    __syncthreads();
    gemm_layer<false, F_W   >(Wb3, b3, bufB, LDB, nullptr, 0, out, row0, n_hr, lane, wave);
}

extern "C" void kernel_launch(void* const* d_in, const int* in_sizes, int n_in,
                              void* d_out, int out_size, void* d_ws, size_t ws_size,
                              hipStream_t stream) {
    const float* v      = (const float*)d_in[0];
    const float* e_rel  = (const float*)d_in[1];
    const int*   idx    = (const int*)  d_in[2];
    const float* v_skip = (const float*)d_in[3];
    const float* We     = (const float*)d_in[4];
    const float* be     = (const float*)d_in[5];
    const float* ln_g   = (const float*)d_in[6];
    const float* ln_b   = (const float*)d_in[7];
    const float* W0 = (const float*)d_in[8];  const float* b0 = (const float*)d_in[9];
    const float* W1 = (const float*)d_in[10]; const float* b1 = (const float*)d_in[11];
    const float* W2 = (const float*)d_in[12]; const float* b2 = (const float*)d_in[13];
    const float* W3 = (const float*)d_in[14]; const float* b3 = (const float*)d_in[15];

    const long long n_hr = in_sizes[1];   // e_rel has N_HR elements

    // bf16 weight staging in workspace (deterministic, rebuilt every call)
    unsigned short* Wb0 = (unsigned short*)d_ws;
    unsigned short* Wb1 = Wb0 + F_W * IN_MSG;
    unsigned short* Wb2 = Wb1 + F_W * F_W;
    unsigned short* Wb3 = Wb2 + F_W * F_W;

    const int n0 = F_W * IN_MSG, n1 = F_W * F_W;
    cvt_bf16_kernel<<<(n0 + 255) / 256, 256, 0, stream>>>(W0, Wb0, n0);
    cvt_bf16_kernel<<<(n1 + 255) / 256, 256, 0, stream>>>(W1, Wb1, n1);
    cvt_bf16_kernel<<<(n1 + 255) / 256, 256, 0, stream>>>(W2, Wb2, n1);
    cvt_bf16_kernel<<<(n1 + 255) / 256, 256, 0, stream>>>(W3, Wb3, n1);

    const size_t smem = (size_t)(TILE_M * LDA + 2 * TILE_M * LDB) * sizeof(unsigned short);
    (void)hipFuncSetAttribute((const void*)meshup_main,
                              hipFuncAttributeMaxDynamicSharedMemorySize, (int)smem);

    dim3 grid((unsigned)((n_hr + TILE_M - 1) / TILE_M));
    meshup_main<<<grid, 256, smem, stream>>>(
        v, e_rel, idx, v_skip, We, be, ln_g, ln_b,
        Wb0, b0, Wb1, b1, Wb2, b2, Wb3, b3,
        (float*)d_out, n_hr);
}
